// EuclideanFastAttention_2061584302408
// MI455X (gfx1250) — matrix operand order repaired and checked by reading, not compiled, not run on hardware
//
#include <hip/hip_runtime.h>

// ---------------- CDNA5 WMMA types & helpers (wave32) ----------------
typedef __attribute__((ext_vector_type(16))) __bf16 v16bf;
typedef __attribute__((ext_vector_type(8)))  float  v8f;
typedef __attribute__((ext_vector_type(8)))  __bf16 bf8;   // 16-byte pack
typedef __attribute__((ext_vector_type(4)))  __bf16 bf4;   // 8-byte pack

#define DEV __device__ __forceinline__

DEV v16bf combine16(bf8 lo, bf8 hi) {
  return __builtin_shufflevector(lo, hi, 0, 1, 2, 3, 4, 5, 6, 7,
                                 8, 9, 10, 11, 12, 13, 14, 15);
}

// A fragment: 16x32 (MxK) bf16, row-major source, row stride ld (elements).
// Per ISA 7.12.2: lane m=lane&15, half=lane>>4 holds K = {8h..8h+7} and {16+8h..+7}
// -> two contiguous 16-byte LDS loads.
DEV v16bf load_A_MK(const __bf16* p, int ld) {
  const int lane = threadIdx.x & 31;
  const __bf16* row = p + (lane & 15) * ld + 8 * (lane >> 4);
  bf8 lo = *(const bf8*)row;          // K = base .. base+7   (VGPR0..3)
  bf8 hi = *(const bf8*)(row + 16);   // K = base+16 .. +23   (VGPR4..7)
  return combine16(lo, hi);
}

// B fragment: 32x16 (KxN) bf16 where the source tile is stored N x K row-major.
// Per lane: col n=lane&15, K = {8h..8h+7} and {16+8h..+7} (same as A).
DEV v16bf load_B_NK(const __bf16* p, int ld) {
  const int lane = threadIdx.x & 31;
  const __bf16* row = p + (lane & 15) * ld + 8 * (lane >> 4);
  bf8 lo = *(const bf8*)row;
  bf8 hi = *(const bf8*)(row + 16);
  return combine16(lo, hi);
}

DEV v8f wmma_bf16(v16bf a, v16bf b, v8f c) {
  return __builtin_amdgcn_wmma_f32_16x16x32_bf16(false, a, false, b, (short)0, c,
                                                 false, false);
}

// ---------------- constants ----------------
#define NN    4096
#define NDEG  9
#define FF    128
#define MDIM  1152        // 9*128
#define GG    6
#define NB    16
#define NPB   256         // nodes per batch

// ============================================================================
// Kernel 0: transpose+quantize weights: WT[mat][l][n][k] = (bf16) W[mat][l][k][n]
// 3*3*128*128 elements. grid = 144 blocks * 256 threads, 4 k-elements each.
// ============================================================================
__global__ __launch_bounds__(256)
void wt_kernel(const float* __restrict__ Wq, const float* __restrict__ Wk,
               const float* __restrict__ Wv, __bf16* __restrict__ WT) {
  const float* Ws[3] = {Wq, Wk, Wv};
  int t = blockIdx.x * 256 + threadIdx.x;       // [0, 3*3*128*32)
  if (t >= 3 * 3 * 128 * 32) return;
  int k4  = (t & 31) * 4;
  int n   = (t >> 5) & 127;
  int r   = t >> 12;                            // [0,9)
  int l   = r % 3;
  int mat = r / 3;
  const float* src = Ws[mat] + (size_t)l * FF * FF;
  bf4 o;
#pragma unroll
  for (int j = 0; j < 4; ++j) o[j] = (__bf16)src[(size_t)(k4 + j) * FF + n];
  *(bf4*)&WT[(((size_t)mat * 3 + l) * FF + n) * FF + k4] = o;
}

// ============================================================================
// Stage A: per-degree dense (q,k,v) + RoPE. Block = (64-node tile, degree m).
// grid = 64 * 9 = 576 blocks, 256 threads (8 waves).
// ============================================================================
__global__ __launch_bounds__(256)
void edense_rope_kernel(const float* __restrict__ inputs,
                        const float* __restrict__ positions,
                        const __bf16* __restrict__ WT,
                        const float* __restrict__ bq, const float* __restrict__ bk,
                        const float* __restrict__ bv,
                        __bf16* __restrict__ qr, __bf16* __restrict__ kr,
                        __bf16* __restrict__ vfT) {
  const int m   = blockIdx.x % NDEG;   // degree index
  const int nb  = blockIdx.x / NDEG;   // 64-node block
  const int l   = (m == 0) ? 0 : ((m < 4) ? 1 : 2);   // DEG_IDX
  const int tid = threadIdx.x;
  const int w = tid >> 5, lane = tid & 31;
  const int col = lane & 15, half = lane >> 4;
  const int rtw = w & 3;               // row tile (16 nodes)
  const int ctb = (w >> 2) * 4;        // col tile base (of 8)

  __shared__ __align__(16) __bf16 Xs[64 * 128];   // nodes x feat (MxK)
  __shared__ __align__(16) __bf16 Wl[128 * 128];  // out-feat x in-feat (NxK)

  // Load X tile (f32 -> bf16, packed 8-byte LDS stores)
  for (int idx = tid; idx < 64 * 128 / 4; idx += 256) {
    int e = idx * 4, i = e >> 7, f = e & 127;
    const float4 xv =
        *(const float4*)(inputs + ((size_t)(nb * 64 + i) * NDEG + (size_t)m) * FF + f);
    bf4 o;
    o[0] = (__bf16)xv.x; o[1] = (__bf16)xv.y; o[2] = (__bf16)xv.z; o[3] = (__bf16)xv.w;
    *(bf4*)&Xs[e] = o;
  }

  const float* bs[3] = {bq, bk, bv};
  __bf16* dsts[2]    = {qr, kr};

  for (int mat = 0; mat < 3; ++mat) {
    __syncthreads();
    const __bf16* Wsrc = WT + ((size_t)mat * 3 + l) * FF * FF;  // N x K, bf16
    for (int idx = tid; idx < 128 * 128 / 8; idx += 256) {
      int e = idx * 8;
      *(bf8*)&Wl[e] = *(const bf8*)&Wsrc[e];
    }
    __syncthreads();

    const v8f zero = {0.f, 0.f, 0.f, 0.f, 0.f, 0.f, 0.f, 0.f};
    v8f acc[4];
#pragma unroll
    for (int i = 0; i < 4; ++i) acc[i] = zero;

#pragma unroll
    for (int ks = 0; ks < 4; ++ks) {
      v16bf a = load_A_MK(&Xs[(rtw * 16) * 128 + ks * 32], 128);
#pragma unroll
      for (int i = 0; i < 4; ++i) {
        v16bf bb = load_B_NK(&Wl[((ctb + i) * 16) * 128 + ks * 32], 128);
        acc[i] = wmma_bf16(a, bb, acc[i]);
      }
    }

    // Post-process & store
    const int node0 = nb * 64 + rtw * 16 + 8 * half;  // 8 consecutive nodes per lane
    for (int i = 0; i < 4; ++i) {
      const int f = (ctb + i) * 16 + col;
      const float biasv = (m == 0) ? bs[mat][f] : 0.0f;
      if (mat == 2) {  // V: no RoPE; vfT[d][node], 8 consecutive nodes -> one 16B store
        bf8 vv;
#pragma unroll
        for (int r = 0; r < 8; ++r) vv[r] = (__bf16)(acc[i][r] + biasv);
        *(bf8*)&vfT[(size_t)(m * FF + f) * NN + node0] = vv;
      } else {
        const int t = f >> 1;
        const bool even = (f & 1) == 0;
        const float theta_t = (8.0f * (float)t / 63.0f) * 0.1f;  // linspace(0,8,64)/10
        __bf16* dst = dsts[mat];
#pragma unroll
        for (int r = 0; r < 8; ++r) {
          int node = node0 + r;
          float self  = acc[i][r] + biasv;
          float other = __shfl_xor(self, 1, 32);      // paired feature (f^1), same row
          float x1 = even ? self : other;             // even-feature value
          float x2 = even ? other : self;             // odd-feature value
          const float* pp = positions + (size_t)node * 3;
          float P[3] = {pp[0], pp[1], pp[2]};
#pragma unroll
          for (int axis = 0; axis < 3; ++axis) {
            float s, c;
            __sincosf(P[axis] * theta_t, &s, &c);
            // grid_u rows: +axis then -axis -> cos same, sin negated
            float v0 = even ? (x1 * c - x2 * s) : ( x1 * s + x2 * c);  // g = 2*axis
            float v1 = even ? (x1 * c + x2 * s) : (-x1 * s + x2 * c);  // g = 2*axis+1
            size_t base = ((size_t)node * GG + 2 * axis) * MDIM + (size_t)m * FF + f;
            dst[base]        = (__bf16)v0;
            dst[base + MDIM] = (__bf16)v1;
          }
        }
      }
    }
  }
}

// ============================================================================
// Stage B1: S[b,g] = (1/6) * Q_g K_g^T  (256x256, K over Mdim=1152).
// Block = (b, g, 64-row tile). grid = 16*6*4 = 384 blocks.
// ============================================================================
__global__ __launch_bounds__(256)
void kv_s_kernel(const __bf16* __restrict__ qr, const __bf16* __restrict__ kr,
                 __bf16* __restrict__ S) {
  const int bx = blockIdx.x;
  const int b  = bx / 24;
  const int g  = (bx % 24) / 4;
  const int rt = bx % 4;
  const int tid = threadIdx.x;
  const int w = tid >> 5, lane = tid & 31;
  const int col = lane & 15, half = lane >> 4;
  const int rtw = w & 3;
  const int ctb = (w >> 2) * 8;   // 16 col tiles total, 8 per wave group

  __shared__ __align__(16) __bf16 Qt[64 * 64];    // node rows x K (MxK)
  __shared__ __align__(16) __bf16 Kt[256 * 64];   // node' rows x K (NxK)

  const v8f zero = {0.f, 0.f, 0.f, 0.f, 0.f, 0.f, 0.f, 0.f};
  v8f acc[8];
#pragma unroll
  for (int i = 0; i < 8; ++i) acc[i] = zero;

  for (int k0 = 0; k0 < MDIM; k0 += 64) {
    for (int idx = tid; idx < 64 * 64 / 8; idx += 256) {
      int e = idx * 8, i = e >> 6, j = e & 63;
      *(bf8*)&Qt[e] =
          *(const bf8*)&qr[(((size_t)(b * NPB + rt * 64 + i)) * GG + g) * MDIM + k0 + j];
    }
    for (int idx = tid; idx < 256 * 64 / 8; idx += 256) {
      int e = idx * 8, i = e >> 6, j = e & 63;
      *(bf8*)&Kt[e] =
          *(const bf8*)&kr[(((size_t)(b * NPB + i)) * GG + g) * MDIM + k0 + j];
    }
    __syncthreads();
#pragma unroll
    for (int kk = 0; kk < 64; kk += 32) {
      v16bf a = load_A_MK(&Qt[(rtw * 16) * 64 + kk], 64);
#pragma unroll
      for (int i = 0; i < 8; ++i) {
        v16bf bb = load_B_NK(&Kt[((ctb + i) * 16) * 64 + kk], 64);
        acc[i] = wmma_bf16(a, bb, acc[i]);
      }
    }
    __syncthreads();
  }

  const float wg = 1.0f / 6.0f;  // grid_w
#pragma unroll
  for (int i = 0; i < 8; ++i) {
    int cc = (ctb + i) * 16 + col;
#pragma unroll
    for (int r = 0; r < 8; ++r) {
      int row = rt * 64 + rtw * 16 + r + 8 * half;
      S[(((size_t)b * GG + g) * NPB + row) * NPB + cc] = (__bf16)(acc[i][r] * wg);
    }
  }
}

// ============================================================================
// Stage B2: out[b] = sum_g S[b,g] @ V[b]; fp32 output with graph mask.
// V is pre-transposed (vfT[d][node]) so B fragments are contiguous.
// Block = (b, 64-row tile, 128-col tile). grid = 16*4*9 = 576 blocks.
// ============================================================================
__global__ __launch_bounds__(256)
void out_kernel(const __bf16* __restrict__ S, const __bf16* __restrict__ vfT,
                const unsigned char* __restrict__ gmask, float* __restrict__ out) {
  const int bx = blockIdx.x;
  const int b  = bx / 36;
  const int nt = (bx % 36) / 9;
  const int dt = bx % 9;
  const int tid = threadIdx.x;
  const int w = tid >> 5, lane = tid & 31;
  const int col = lane & 15, half = lane >> 4;
  const int rtw = w & 3;
  const int ctb = (w >> 2) * 4;   // 8 col tiles total, 4 per wave group

  __shared__ __align__(16) __bf16 St[64 * 64];    // out-node rows x K=node' (MxK)
  __shared__ __align__(16) __bf16 Vt[128 * 64];   // d rows x K=node' (NxK)

  const v8f zero = {0.f, 0.f, 0.f, 0.f, 0.f, 0.f, 0.f, 0.f};
  v8f acc[4];
#pragma unroll
  for (int i = 0; i < 4; ++i) acc[i] = zero;

  for (int g = 0; g < GG; ++g) {
    for (int c0 = 0; c0 < NPB; c0 += 64) {
      for (int idx = tid; idx < 64 * 64 / 8; idx += 256) {
        int e = idx * 8, i = e >> 6, j = e & 63;
        *(bf8*)&St[e] =
            *(const bf8*)&S[(((size_t)b * GG + g) * NPB + nt * 64 + i) * NPB + c0 + j];
      }
      for (int idx = tid; idx < 128 * 64 / 8; idx += 256) {
        int e = idx * 8, d = e >> 6, j = e & 63;
        *(bf8*)&Vt[e] =
            *(const bf8*)&vfT[(size_t)(dt * FF + d) * NN + b * NPB + c0 + j];
      }
      __syncthreads();
#pragma unroll
      for (int kk = 0; kk < 64; kk += 32) {
        v16bf a = load_A_MK(&St[(rtw * 16) * 64 + kk], 64);
#pragma unroll
        for (int i = 0; i < 4; ++i) {
          v16bf bb = load_B_NK(&Vt[((ctb + i) * 16) * 64 + kk], 64);
          acc[i] = wmma_bf16(a, bb, acc[i]);
        }
      }
      __syncthreads();
    }
  }

  const float msk = gmask[b] ? 1.0f : 0.0f;
#pragma unroll
  for (int i = 0; i < 4; ++i) {
    int d = dt * FF + (ctb + i) * 16 + col;
#pragma unroll
    for (int r = 0; r < 8; ++r) {
      int node = b * NPB + nt * 64 + rtw * 16 + r + 8 * half;
      out[(size_t)node * MDIM + d] = acc[i][r] * msk;
    }
  }
}

// ============================================================================
extern "C" void kernel_launch(void* const* d_in, const int* in_sizes, int n_in,
                              void* d_out, int out_size, void* d_ws, size_t ws_size,
                              hipStream_t stream) {
  (void)in_sizes; (void)n_in; (void)out_size; (void)ws_size;
  const float* inputs    = (const float*)d_in[0];
  const float* positions = (const float*)d_in[1];
  // d_in[2] = batch_segments (repeat(arange(16), 256)) -> b = node/256, unused
  const unsigned char* gmask = (const unsigned char*)d_in[3];
  const float* Wq = (const float*)d_in[4];
  const float* bq = (const float*)d_in[5];
  const float* Wk = (const float*)d_in[6];
  const float* bk = (const float*)d_in[7];
  const float* Wv = (const float*)d_in[8];
  const float* bv = (const float*)d_in[9];

  // Workspace (bf16): qr, kr (N*6*1152), vfT (1152*N), S (16*6*256*256), WT (3*3*128*128)
  const size_t QR = (size_t)NN * GG * MDIM;
  __bf16* qr  = (__bf16*)d_ws;
  __bf16* kr  = qr + QR;
  __bf16* vfT = kr + QR;
  __bf16* S   = vfT + (size_t)NN * MDIM;
  __bf16* WT  = S + (size_t)NB * GG * NPB * NPB;

  wt_kernel<<<144, 256, 0, stream>>>(Wq, Wk, Wv, WT);
  edense_rope_kernel<<<(NN / 64) * NDEG, 256, 0, stream>>>(
      inputs, positions, WT, bq, bk, bv, qr, kr, vfT);
  kv_s_kernel<<<NB * GG * 4, 256, 0, stream>>>(qr, kr, S);
  out_kernel<<<NB * 4 * 9, 256, 0, stream>>>(S, vfT, gmask, (float*)d_out);
}
